// TemplatePointwiseAttention_1279900254325
// MI455X (gfx1250) — matrix operand order repaired and checked by reading, not compiled, not run on hardware
//
#include <hip/hip_runtime.h>

typedef __attribute__((ext_vector_type(16))) _Float16 v16h;
typedef __attribute__((ext_vector_type(8)))  _Float16 v8h;
typedef __attribute__((ext_vector_type(8)))  float    v8f;

#define NROW   (384 * 384)     // flattened (i,j) rows
#define NT_TPL 4               // templates
#define CZ     128
#define NTILES (NROW / 16)     // 9216 16-row tiles

// LDS strides (elements), padded to kill bank conflicts
#define WLDK 136               // 128-wide f16 rows: 272B = 68 words -> 4-word skew per row
#define OLDK 72                // 64-wide f16 rows: 144B = 36 words
#define SLDK 132               // f32 stage rows: 132 words -> 4-word skew (528B, 16B-mult)

// ---- A fragment (16x32 f16, row-major source [16][ldk]) ----
// lane<16: M=lane,    K = k0+{0..7, 16..23}
// lane>=16: M=lane-16, K = k0+{8..15, 24..31}
__device__ __forceinline__ v16h ldsA16x32(const _Float16* tile, int ldk, int k0, int lane) {
  const int m    = lane & 15;
  const int koff = k0 + ((lane & 16) ? 8 : 0);
  const _Float16* p = tile + m * ldk + koff;
  v8h lo = *(const v8h*)(p);
  v8h hi = *(const v8h*)(p + 16);
  v16h r;
#pragma unroll
  for (int i = 0; i < 8; ++i) { r[i] = lo[i]; r[8 + i] = hi[i]; }
  return r;
}

// ---- B fragment (32x16 f16, from transposed weights [N][ldk]) ----
// lane<16: N=lane, K=k0+{0..7, 16..23} ; lane>=16: N=lane-16, K=k0+{8..15, 24..31}
__device__ __forceinline__ v16h ldsB32x16(const _Float16* wT, int ldk, int k0, int lane) {
  const int n    = lane & 15;
  const int koff = k0 + ((lane & 16) ? 8 : 0);
  const _Float16* p = wT + n * ldk + koff;
  v8h lo = *(const v8h*)(p);
  v8h hi = *(const v8h*)(p + 16);
  v16h r;
#pragma unroll
  for (int i = 0; i < 8; ++i) { r[i] = lo[i]; r[8 + i] = hi[i]; }
  return r;
}

// ---- async global -> LDS stage: 16 rows x 128 f32 (8KB contiguous), b128 per lane ----
__device__ __forceinline__ void issue_stage_async(const float* __restrict__ src,
                                                  const float* stageBuf, int tid) {
  const unsigned ldsBase = (unsigned)(uintptr_t)stageBuf;   // low 32 bits = LDS offset
  const unsigned long long gb = (unsigned long long)(uintptr_t)src;  // uniform -> SGPR pair
#pragma unroll
  for (int i = 0; i < 4; ++i) {
    const int it   = tid + i * 128;                 // 0..511 (float4 index)
    const unsigned lds = ldsBase + (unsigned)((it >> 5) * (SLDK * 4) + (it & 31) * 16);
    const int voff = it * 16;                       // byte offset into 8KB chunk
    asm volatile("global_load_async_to_lds_b128 %0, %1, %2"
                 :
                 : "v"(lds), "v"(voff), "s"(gb)
                 : "memory");
  }
}

__device__ __forceinline__ void wait_async0() {
  asm volatile("s_wait_asynccnt 0" ::: "memory");
}

// ---- LayerNorm 16x128 f32 stage tile -> f16 tile [16][WLDK] (no internal syncs) ----
__device__ __forceinline__ void ln_from_stage(const float* stageBuf, _Float16* dstT,
                                              const float* g, const float* b, int tid) {
  const int r = tid >> 3, part = tid & 7;           // 8 lanes per row
  const float* row = stageBuf + r * SLDK + part * 16;
  float s = 0.f, ss = 0.f;
#pragma unroll
  for (int c = 0; c < 16; ++c) { const float x = row[c]; s += x; ss += x * x; }
  s += __shfl_xor(s, 1); ss += __shfl_xor(ss, 1);
  s += __shfl_xor(s, 2); ss += __shfl_xor(ss, 2);
  s += __shfl_xor(s, 4); ss += __shfl_xor(ss, 4);
  const float mean = s * (1.f / 128.f);
  const float var  = ss * (1.f / 128.f) - mean * mean;
  const float rstd = rsqrtf(var + 1e-5f);
  _Float16* drow = dstT + r * WLDK + part * 16;
  const int cb = part * 16;
#pragma unroll
  for (int c = 0; c < 16; ++c) {
    const float y = (row[c] - mean) * rstd * g[cb + c] + b[cb + c];
    drow[c] = (_Float16)y;
  }
}

extern "C" __global__ void __launch_bounds__(128)
tpa_fused(const float* __restrict__ tmpl_feat, const float* __restrict__ pair,
          const float* __restrict__ ln_pair_g, const float* __restrict__ ln_pair_b,
          const float* __restrict__ ln_tmpl_g, const float* __restrict__ ln_tmpl_b,
          const float* __restrict__ Wq, const float* __restrict__ Wk,
          const float* __restrict__ Wv, const float* __restrict__ Wg,
          const float* __restrict__ bg, const float* __restrict__ Wo,
          const float* __restrict__ bo, float* __restrict__ out) {
  extern __shared__ char smem[];
  _Float16* WqT   = (_Float16*)smem;          // [64][WLDK]
  _Float16* WkT   = WqT + 64 * WLDK;
  _Float16* WvT   = WkT + 64 * WLDK;
  _Float16* WgT   = WvT + 64 * WLDK;
  _Float16* WoT   = WgT + 64 * WLDK;          // [128][OLDK]
  _Float16* pairT = WoT + 128 * OLDK;         // [16][WLDK]
  _Float16* tmplT = pairT + 16 * WLDK;        // [4][16][WLDK]
  _Float16* valsT = tmplT + 4 * 16 * WLDK;    // [16][OLDK]
  float* stage = (float*)(valsT + 16 * OLDK); // [5][16][SLDK]  (0=pair, 1..4=templates)
  float* gpS = stage + 5 * 16 * SLDK;
  float* bpS = gpS + 128;
  float* gtS = bpS + 128;
  float* btS = gtS + 128;
  float* boS = btS + 128;
  float* bgS = boS + 128;

  const int tid  = threadIdx.x;
  const int wv   = tid >> 5;
  const int lane = tid & 31;

  // ---- once per block: weights -> f16, transposed into LDS ----
  for (int idx = tid; idx < 128 * 64; idx += 128) {
    const int k = idx >> 6, n = idx & 63;
    const int o = n * WLDK + k;
    WqT[o] = (_Float16)Wq[idx];
    WkT[o] = (_Float16)Wk[idx];
    WvT[o] = (_Float16)Wv[idx];
    WgT[o] = (_Float16)Wg[idx];
  }
  for (int idx = tid; idx < 64 * 128; idx += 128) {
    const int k = idx >> 7, n = idx & 127;
    WoT[n * OLDK + k] = (_Float16)Wo[idx];
  }
  gpS[tid] = ln_pair_g[tid]; bpS[tid] = ln_pair_b[tid];
  gtS[tid] = ln_tmpl_g[tid]; btS[tid] = ln_tmpl_b[tid];
  boS[tid] = bo[tid];
  if (tid < 64) bgS[tid] = bg[tid];

  // ---- pipeline prologue: async-stage the first tile ----
  {
    const int t0 = blockIdx.x;
    if (t0 < NTILES) {
      const size_t R0 = (size_t)t0 * 16;
      issue_stage_async(pair + R0 * CZ, stage, tid);
#pragma unroll
      for (int t = 0; t < NT_TPL; ++t)
        issue_stage_async(tmpl_feat + ((size_t)t * NROW + R0) * CZ,
                          stage + (1 + t) * 16 * SLDK, tid);
    }
  }
  __syncthreads();   // weights visible to all

  for (int tile = blockIdx.x; tile < NTILES; tile += gridDim.x) {
    const int R0 = tile * 16;

    // ---- Phase A: wait for this tile's async-staged data, then LN -> f16 LDS ----
    wait_async0();
    __syncthreads();                                   // (a) stage complete block-wide
    ln_from_stage(stage, pairT, gpS, bpS, tid);
#pragma unroll
    for (int t = 0; t < NT_TPL; ++t)
      ln_from_stage(stage + (1 + t) * 16 * SLDK, tmplT + t * 16 * WLDK, gtS, btS, tid);
    __syncthreads();                                   // (b) f16 tiles ready, stage consumed

    // ---- prefetch next tile via async DMA (overlaps with all compute below) ----
    const int nxt = tile + gridDim.x;
    if (nxt < NTILES) {
      const size_t Rn = (size_t)nxt * 16;
      issue_stage_async(pair + Rn * CZ, stage, tid);
#pragma unroll
      for (int t = 0; t < NT_TPL; ++t)
        issue_stage_async(tmpl_feat + ((size_t)t * NROW + Rn) * CZ,
                          stage + (1 + t) * 16 * SLDK, tid);
    }

    // ---- Phase B: wave wv owns head h = wv (D=16 == WMMA N-tile) ----
    const int h = wv;
    const _Float16* WqTh = WqT + h * 16 * WLDK;
    const _Float16* WkTh = WkT + h * 16 * WLDK;
    const _Float16* WvTh = WvT + h * 16 * WLDK;
    const _Float16* WgTh = WgT + h * 16 * WLDK;

    v8f qacc = {}; v8f gacc = {};
#pragma unroll
    for (int k0 = 0; k0 < 4; ++k0) {
      const v16h a = ldsA16x32(pairT, WLDK, k0 * 32, lane);
      qacc = __builtin_amdgcn_wmma_f32_16x16x32_f16(false, a, false,
                 ldsB32x16(WqTh, WLDK, k0 * 32, lane), (short)0, qacc, false, false);
      gacc = __builtin_amdgcn_wmma_f32_16x16x32_f16(false, a, false,
                 ldsB32x16(WgTh, WLDK, k0 * 32, lane), (short)0, gacc, false, false);
    }
    const float bgv = bgS[h * 16 + (lane & 15)];
    float gsig[8];
#pragma unroll
    for (int r = 0; r < 8; ++r) gsig[r] = 1.f / (1.f + __expf(-(gacc[r] + bgv)));

    // scores[t][m]: elementwise Q*K + 16-lane butterfly (result replicated over d-lanes)
    float sc[4][8];
#pragma unroll
    for (int t = 0; t < NT_TPL; ++t) {
      v8f kacc = {};
#pragma unroll
      for (int k0 = 0; k0 < 4; ++k0)
        kacc = __builtin_amdgcn_wmma_f32_16x16x32_f16(false,
                   ldsA16x32(tmplT + t * 16 * WLDK, WLDK, k0 * 32, lane), false,
                   ldsB32x16(WkTh, WLDK, k0 * 32, lane), (short)0, kacc, false, false);
#pragma unroll
      for (int r = 0; r < 8; ++r) {
        float p = qacc[r] * kacc[r];
        p += __shfl_xor(p, 1); p += __shfl_xor(p, 2);
        p += __shfl_xor(p, 4); p += __shfl_xor(p, 8);
        sc[t][r] = p * 0.25f;                          // 1/sqrt(D), D=16
      }
    }
    // softmax over t=4, purely elementwise (replicated across d-lanes)
    float w[4][8];
#pragma unroll
    for (int r = 0; r < 8; ++r) {
      const float m = fmaxf(fmaxf(sc[0][r], sc[1][r]), fmaxf(sc[2][r], sc[3][r]));
      const float e0 = __expf(sc[0][r] - m), e1 = __expf(sc[1][r] - m);
      const float e2 = __expf(sc[2][r] - m), e3 = __expf(sc[3][r] - m);
      const float inv = 1.f / (e0 + e1 + e2 + e3);
      w[0][r] = e0 * inv; w[1][r] = e1 * inv; w[2][r] = e2 * inv; w[3][r] = e3 * inv;
    }
    // weighted V sum, gated
    v8f vals = {};
#pragma unroll
    for (int t = 0; t < NT_TPL; ++t) {
      v8f vacc = {};
#pragma unroll
      for (int k0 = 0; k0 < 4; ++k0)
        vacc = __builtin_amdgcn_wmma_f32_16x16x32_f16(false,
                   ldsA16x32(tmplT + t * 16 * WLDK, WLDK, k0 * 32, lane), false,
                   ldsB32x16(WvTh, WLDK, k0 * 32, lane), (short)0, vacc, false, false);
#pragma unroll
      for (int r = 0; r < 8; ++r) vals[r] += w[t][r] * vacc[r];
    }
    const int colh = h * 16 + (lane & 15);
    const int mhi  = (lane >> 4) * 8;                  // C-layout: M = r + 8*(lane>>4)
#pragma unroll
    for (int r = 0; r < 8; ++r)
      valsT[(mhi + r) * OLDK + colh] = (_Float16)(vals[r] * gsig[r]);
    __syncthreads();                                   // (c) valsT ready

    // ---- Phase C: out = vals @ Wo + bo (each wave: 2 of 8 N-tiles) ----
    const v16h a0 = ldsA16x32(valsT, OLDK, 0,  lane);
    const v16h a1 = ldsA16x32(valsT, OLDK, 32, lane);
#pragma unroll
    for (int ntl = 0; ntl < 2; ++ntl) {
      const int nt = wv + ntl * 4;
      v8f acc = {};
      acc = __builtin_amdgcn_wmma_f32_16x16x32_f16(false, a0, false,
                ldsB32x16(WoT + nt * 16 * OLDK, OLDK, 0, lane), (short)0, acc, false, false);
      acc = __builtin_amdgcn_wmma_f32_16x16x32_f16(false, a1, false,
                ldsB32x16(WoT + nt * 16 * OLDK, OLDK, 32, lane), (short)0, acc, false, false);
      const int col = nt * 16 + (lane & 15);
      const float bov = boS[col];
      float* orow = out + (size_t)(R0 + mhi) * CZ + col;
#pragma unroll
      for (int r = 0; r < 8; ++r) orow[(size_t)r * CZ] = acc[r] + bov;
    }
    // no barrier needed here: top-of-loop (a) orders valsT/tmplT reuse
  }
}

extern "C" void kernel_launch(void* const* d_in, const int* in_sizes, int n_in,
                              void* d_out, int out_size, void* d_ws, size_t ws_size,
                              hipStream_t stream) {
  (void)in_sizes; (void)n_in; (void)out_size; (void)d_ws; (void)ws_size;
  const float* tmpl_feat = (const float*)d_in[0];
  const float* pair      = (const float*)d_in[1];
  const float* ln_pair_g = (const float*)d_in[2];
  const float* ln_pair_b = (const float*)d_in[3];
  const float* ln_tmpl_g = (const float*)d_in[4];
  const float* ln_tmpl_b = (const float*)d_in[5];
  const float* Wq        = (const float*)d_in[6];
  const float* Wk        = (const float*)d_in[7];
  const float* Wv        = (const float*)d_in[8];
  const float* Wg        = (const float*)d_in[9];
  const float* bg        = (const float*)d_in[10];
  const float* Wo        = (const float*)d_in[11];
  const float* bo        = (const float*)d_in[12];
  float* out = (float*)d_out;

  // LDS: f16 weights+tiles 112128B + 5 async stage buffers 42240B + params 2816B = 157184B
  // -> 2 blocks per 320KB WGP
  const size_t shmem = (size_t)(64 * WLDK * 4 + 128 * OLDK + 16 * WLDK * 5 + 16 * OLDK) * 2
                     + (size_t)(5 * 16 * SLDK + 128 * 5 + 64) * 4;
  dim3 grid(1024), block(128);
  tpa_fused<<<grid, block, shmem, stream>>>(tmpl_feat, pair, ln_pair_g, ln_pair_b,
                                            ln_tmpl_g, ln_tmpl_b, Wq, Wk, Wv, Wg,
                                            bg, Wo, bo, out);
}
